// AggregatedAttention_24799141167814
// MI455X (gfx1250) — matrix operand order repaired and checked by reading, not compiled, not run on hardware
//
#include <hip/hip_runtime.h>

#define DEV __device__ __forceinline__
typedef unsigned short u16;
typedef unsigned int u32;
typedef __attribute__((ext_vector_type(16))) __bf16 v16bf;
typedef __attribute__((ext_vector_type(8)))  float  v8f;

// ---------------- problem constants ----------------
constexpr int BATCH = 8;
constexpr int NTOK  = 3136;   // 56*56
constexpr int DIMC  = 256;
constexpr int NHEAD = 8;
constexpr int HD    = 32;
constexpr int PLEN  = 49;     // 7*7 pooled tokens
constexpr int HW    = 56;
constexpr int M1    = BATCH * NTOK;   // 25088 rows
constexpr int NCAT  = 1024;           // q(256) | k(256) | v(256) | sr(256)
constexpr float NEGV = -1.0e9f;

// ---------------- workspace layout (bytes) ----------------
constexpr size_t SZ_XBF  = (size_t)M1 * DIMC * 2;          // x in bf16
constexpr size_t SZ_WCAT = (size_t)DIMC * NCAT * 2;        // concat weight bf16 (stored [N][K])
constexpr size_t SZ_BIA  = (size_t)NCAT * 4;               // concat bias f32
constexpr size_t SZ_PRJ  = (size_t)DIMC * DIMC * 2;        // proj weight bf16 (stored [N][K])
constexpr size_t SZ_TAB  = (size_t)4096 * 8 * 4;           // cpb table f32
constexpr size_t SZ_XPL  = (size_t)BATCH * PLEN * DIMC * 4;// pooled+LN f32
constexpr size_t SZ_KP   = (size_t)BATCH * NHEAD * 64 * 32 * 2; // k_pool bf16 [b][h][64][32]
constexpr size_t SZ_VPT  = SZ_KP;                          // v_pool^T bf16 [b][h][32][64]
constexpr size_t SZ_HND2 = (size_t)BATCH * NHEAD * NTOK * HD * 2; // bf16 per-head maps
constexpr size_t SZ_HND4 = (size_t)BATCH * NHEAD * NTOK * HD * 4;

constexpr size_t OFF_XBF  = 0;
constexpr size_t OFF_WCAT = OFF_XBF  + SZ_XBF;
constexpr size_t OFF_BIA  = OFF_WCAT + SZ_WCAT;
constexpr size_t OFF_PRJ  = OFF_BIA  + SZ_BIA;
constexpr size_t OFF_TAB  = OFF_PRJ  + SZ_PRJ;
constexpr size_t OFF_XPL  = OFF_TAB  + SZ_TAB;
constexpr size_t OFF_KP   = OFF_XPL  + SZ_XPL;
constexpr size_t OFF_VPT  = OFF_KP   + SZ_KP;
constexpr size_t OFF_QS   = OFF_VPT  + SZ_VPT;
constexpr size_t OFF_QN   = OFF_QS   + SZ_HND2;
constexpr size_t OFF_KBF  = OFF_QN   + SZ_HND2;
constexpr size_t OFF_VF   = OFF_KBF  + SZ_HND2;
constexpr size_t OFF_OUT1 = OFF_VF   + SZ_HND4;            // 25088 x 1024 f32
constexpr size_t OFF_YBF  = OFF_OUT1;                      // alias: out1 dead when y written

// ---------------- helpers ----------------
DEV u16 f2bf(float f) {
  union { float f; u32 u; } x; x.f = f;
  u32 r = x.u + 0x7FFFu + ((x.u >> 16) & 1u);
  return (u16)(r >> 16);
}
DEV float bf2f(u16 b) {
  union { u32 u; float f; } x; x.u = ((u32)b) << 16;
  return x.f;
}
DEV float wave_sum(float v) {
  #pragma unroll
  for (int o = 16; o > 0; o >>= 1) v += __shfl_xor(v, o);
  return v;
}
DEV float gelu_exact(float x) { return 0.5f * x * (1.0f + erff(x * 0.70710678118654752f)); }

// CDNA5 async DMA: global -> LDS, 16B per lane, tracked by ASYNCcnt.
// lds  = per-lane LDS byte address (low 32 bits of generic shared pointer)
// goff = per-lane 32-bit byte offset from SGPR base (GVS mode)
DEV void async_ld_b128(u32 lds, u32 goff, const void* base) {
  asm volatile("global_load_async_to_lds_b128 %0, %1, %2"
               :: "v"(lds), "v"(goff), "s"(base) : "memory");
}
DEV void wait_async0() { asm volatile("s_wait_asynccnt 0x0" ::: "memory"); }

// A fragment: 16x32 bf16, lane L -> row = row0 + (L%16),
// elems 0..7 = K[k0+8h .. +7], elems 8..15 = K[k0+16+8h .. +7]  (h = L/16)
DEV v16bf load_frag_A(const u16* sA, int ld, int row0, int k0, int lane) {
  union { v16bf v; int4 q[2]; } u;
  const u16* p = sA + (size_t)(row0 + (lane & 15)) * ld + k0 + 8 * (lane >> 4);
  u.q[0] = *(const int4*)p;
  u.q[1] = *(const int4*)(p + 16);
  return u.v;
}
// B fragment: 32x16 bf16 from N-major LDS (sB[n][k]): lane L -> col n0+(L%16),
// elems 0..7 = K[k0+8h .. +7], elems 8..15 = K[k0+16+8h .. +7]  (h = L/16)
DEV v16bf load_frag_B(const u16* sB, int ld, int n0, int k0, int lane) {
  union { v16bf v; int4 q[2]; } u;
  const u16* p = sB + (size_t)(n0 + (lane & 15)) * ld + k0 + 8 * (lane >> 4);
  u.q[0] = *(const int4*)p;
  u.q[1] = *(const int4*)(p + 16);
  return u.v;
}

// ---------------- kernel 0: pack weights (transposed to [N][K]) / zero pads ----------------
__global__ void k_pack(const float* __restrict__ qw, const float* __restrict__ kvw,
                       const float* __restrict__ srw, const float* __restrict__ qb,
                       const float* __restrict__ kvb, const float* __restrict__ srb,
                       const float* __restrict__ projw,
                       u16* __restrict__ wcatT, float* __restrict__ biac,
                       u16* __restrict__ prjT, u16* __restrict__ kp, u16* __restrict__ vpt) {
  int e = blockIdx.x * 256 + threadIdx.x;          // 0 .. 262143
  int k = e >> 10, j = e & 1023;
  float w = (j < 256) ? qw[k * 256 + j]
          : (j < 768) ? kvw[k * 512 + (j - 256)]
                      : srw[k * 256 + (j - 768)];
  wcatT[j * 256 + k] = f2bf(w);                    // [n][k] for async B tiles
  if (e < 65536) prjT[(e & 255) * 256 + (e >> 8)] = f2bf(projw[e]);
  if (e < 1024)
    biac[e] = (e < 256) ? qb[e] : (e < 768) ? kvb[e - 256] : srb[e - 768];
  if (e < (int)(BATCH * NHEAD * 64 * 32)) { kp[e] = 0; vpt[e] = 0; }
}

// ---------------- kernel 1: x -> bf16 ----------------
__global__ void k_cvt_x(const float* __restrict__ x, u16* __restrict__ xbf) {
  int e = blockIdx.x * 256 + threadIdx.x;
  xbf[e] = f2bf(x[e]);
}

// ---------------- kernel 2: bf16 WMMA GEMM  C[M x N] = A[M x 256] * B^T[N x 256] + bias ----
// block 256 threads = 8 waves; tile 128x128; wave = 32(M) x 64(N) = 2x4 wmma tiles.
// Tiles staged with global_load_async_to_lds_b128 (ASYNCcnt), double-buffered LDS.
__global__ __launch_bounds__(256) void k_gemm_bf16(const u16* __restrict__ A,
                                                   const u16* __restrict__ BT,
                                                   const float* __restrict__ bias,
                                                   float* __restrict__ C, int N) {
  __shared__ alignas(16) u16 sA[2][128 * 56];
  __shared__ alignas(16) u16 sB[2][128 * 56];
  const int t = threadIdx.x;
  const int lane = t & 31;
  const int wave = t >> 5;
  const int wm = wave >> 1, wn = wave & 1;
  const int m0 = blockIdx.x * 128;
  const int n0 = blockIdx.y * 128;

  // per-thread staging slots: row r, elem chunks c0 and c0+16 (8 bf16 = 16B each)
  const int r  = t >> 1;
  const int c0 = (t & 1) * 8;
  u32 lA[2], lB[2];
  lA[0] = (u32)(unsigned long long)&sA[0][r * 56 + c0];
  lA[1] = (u32)(unsigned long long)&sA[1][r * 56 + c0];
  lB[0] = (u32)(unsigned long long)&sB[0][r * 56 + c0];
  lB[1] = (u32)(unsigned long long)&sB[1][r * 56 + c0];
  const u32 gA0 = (u32)(((size_t)(m0 + r) * 256 + c0) * 2);  // byte offset in A
  const u32 gB0 = (u32)(((size_t)(n0 + r) * 256 + c0) * 2);  // byte offset in BT

  v8f acc[2][4];
  #pragma unroll
  for (int i = 0; i < 2; i++)
    #pragma unroll
    for (int j = 0; j < 4; j++) acc[i][j] = (v8f){0.f,0.f,0.f,0.f,0.f,0.f,0.f,0.f};

  // prologue: async-load K-tile 0 into buffer 0
  async_ld_b128(lA[0],      gA0,      A);
  async_ld_b128(lA[0] + 32, gA0 + 32, A);
  async_ld_b128(lB[0],      gB0,      BT);
  async_ld_b128(lB[0] + 32, gB0 + 32, BT);

  for (int kt = 0; kt < 8; kt++) {
    const int cur = kt & 1;
    wait_async0();          // my tile-kt chunks (and nothing newer) are in LDS
    __syncthreads();        // everyone's chunks landed; prior compute done
    if (kt < 7) {           // overlap next DMA with this tile's WMMAs
      const u32 kb = (u32)((kt + 1) * 64);   // 32 elems * 2B
      async_ld_b128(lA[cur ^ 1],      gA0 + kb,      A);
      async_ld_b128(lA[cur ^ 1] + 32, gA0 + kb + 32, A);
      async_ld_b128(lB[cur ^ 1],      gB0 + kb,      BT);
      async_ld_b128(lB[cur ^ 1] + 32, gB0 + kb + 32, BT);
    }
    v16bf afr0 = load_frag_A(sA[cur], 56, wm * 32, 0, lane);
    v16bf afr1 = load_frag_A(sA[cur], 56, wm * 32 + 16, 0, lane);
    #pragma unroll
    for (int ni = 0; ni < 4; ni++) {
      v16bf bfr = load_frag_B(sB[cur], 56, wn * 64 + ni * 16, 0, lane);
      acc[0][ni] = __builtin_amdgcn_wmma_f32_16x16x32_bf16(false, afr0, false, bfr,
                                                           (short)0, acc[0][ni], false, false);
      acc[1][ni] = __builtin_amdgcn_wmma_f32_16x16x32_bf16(false, afr1, false, bfr,
                                                           (short)0, acc[1][ni], false, false);
    }
    __syncthreads();        // done reading cur before it is overwritten at kt+2
  }

  const int hl = lane >> 4, ln = lane & 15;
  #pragma unroll
  for (int mi = 0; mi < 2; mi++)
    #pragma unroll
    for (int ni = 0; ni < 4; ni++) {
      int col = n0 + wn * 64 + ni * 16 + ln;
      float bv = bias[col];
      #pragma unroll
      for (int g = 0; g < 8; g++) {
        int row = m0 + wm * 32 + mi * 16 + g + 8 * hl;
        C[(size_t)row * N + col] = acc[mi][ni][g] + bv;
      }
    }
}

// ---------------- kernel 3: epilogue of GEMM1 (q/k/v parts) ----------------
// one block per (b,n) row; wave == head (wave32), lane == head-dim
__global__ void k_epilogue1(const float* __restrict__ out1,
                            const float* __restrict__ qe, const float* __restrict__ temp,
                            const float* __restrict__ slsp,
                            u16* __restrict__ qsb, u16* __restrict__ qnb,
                            u16* __restrict__ kbf, float* __restrict__ vf) {
  int row = blockIdx.x;
  int b = row / NTOK, n = row % NTOK;
  int h = threadIdx.x >> 5, d = threadIdx.x & 31;
  const float* o1 = out1 + (size_t)row * NCAT;
  float spt = log1pf(expf(temp[h])) * slsp[0];

  float qv = o1[h * 32 + d];
  float qn = qv / fmaxf(sqrtf(wave_sum(qv * qv)), 1e-12f);
  float qs = (qn + qe[h * 32 + d]) * spt;
  size_t base = ((size_t)(b * NHEAD + h) * NTOK + n) * HD + d;
  qsb[base] = f2bf(qs);
  qnb[base] = f2bf(qn);

  float kv = o1[256 + h * 32 + d];
  kbf[base] = f2bf(kv / fmaxf(sqrtf(wave_sum(kv * kv)), 1e-12f));
  vf[base] = o1[512 + h * 32 + d];
}

// ---------------- kernel 4: gelu + 8x8 avg pool + LayerNorm (deterministic) ----------------
__global__ void k_poolln(const float* __restrict__ out1, const float* __restrict__ ng,
                         const float* __restrict__ nb, float* __restrict__ xpln) {
  __shared__ float red[256];
  int blk = blockIdx.x;                 // b*49 + p
  int b = blk / PLEN, p = blk % PLEN;
  int ph = p / 7, pw = p % 7;
  int c = threadIdx.x;
  float s = 0.f;
  for (int yy = 0; yy < 8; yy++)
    for (int xx = 0; xx < 8; xx++) {
      int n = (ph * 8 + yy) * HW + pw * 8 + xx;
      s += gelu_exact(out1[((size_t)(b * NTOK + n)) * NCAT + 768 + c]);
    }
  s *= (1.0f / 64.0f);
  red[c] = s; __syncthreads();
  for (int st = 128; st > 0; st >>= 1) { if (c < st) red[c] += red[c + st]; __syncthreads(); }
  float mean = red[0] * (1.0f / 256.0f);
  __syncthreads();
  float dv = s - mean;
  red[c] = dv * dv; __syncthreads();
  for (int st = 128; st > 0; st >>= 1) { if (c < st) red[c] += red[c + st]; __syncthreads(); }
  float var = red[0] * (1.0f / 256.0f);
  xpln[(size_t)blk * 256 + c] = dv * rsqrtf(var + 1e-5f) * ng[c] + nb[c];
}

// ---------------- kernel 5: pooled kv + l2norm, lay out for WMMA B fragments ------------
__global__ void k_kvpool(const float* __restrict__ xpln, const float* __restrict__ kvw,
                         const float* __restrict__ kvb, u16* __restrict__ kp,
                         u16* __restrict__ vpt) {
  __shared__ float sx[256];
  int blk = blockIdx.x;                 // b*49 + p
  int b = blk / PLEN, p = blk % PLEN;
  int t = threadIdx.x;
  sx[t] = xpln[(size_t)blk * 256 + t];
  __syncthreads();
  float a0 = 0.f, a1 = 0.f;
  for (int c = 0; c < 256; c++) {
    float xv = sx[c];
    a0 += xv * kvw[c * 512 + t];
    a1 += xv * kvw[c * 512 + t + 256];
  }
  a0 += kvb[t];
  a1 += kvb[t + 256];
  int h = t >> 5, d = t & 31;           // wave == head again
  float kn = a0 / fmaxf(sqrtf(wave_sum(a0 * a0)), 1e-12f);
  kp[((size_t)(b * NHEAD + h) * 64 + p) * 32 + d] = f2bf(kn);     // [b][h][p(64)][d]
  vpt[((size_t)(b * NHEAD + h) * 32 + d) * 64 + p] = f2bf(a1);    // [b][h][d][p(64)]
}

// ---------------- kernel 6: CPB MLP table (4096 x 8) ----------------
__global__ void k_cpb(const float* __restrict__ rct, const float* __restrict__ fc1w,
                      const float* __restrict__ fc1b, const float* __restrict__ fc2w,
                      const float* __restrict__ fc2b, float* __restrict__ tab) {
  int i = blockIdx.x * 8 + (threadIdx.x >> 5);  // one wave per table row
  int l = threadIdx.x & 31;
  float c0 = rct[i * 2], c1 = rct[i * 2 + 1];
  float acc[8] = {0, 0, 0, 0, 0, 0, 0, 0};
  for (int j = l; j < 512; j += 32) {
    float hv = fmaxf(c0 * fc1w[j] + c1 * fc1w[512 + j] + fc1b[j], 0.f);
    #pragma unroll
    for (int h = 0; h < 8; h++) acc[h] += hv * fc2w[j * 8 + h];
  }
  #pragma unroll
  for (int h = 0; h < 8; h++) {
    float s = wave_sum(acc[h]);
    if (l == 0) tab[i * 8 + h] = s + fc2b[h];
  }
}

// ---------------- kernel 7: fused attention (64-row tile per (b,h)) ----------------
// 4 waves: WMMA attn_pool, VALU local window, LDS softmax(58), WMMA x_pool, VALU x_local
__global__ __launch_bounds__(128) void k_attn(const u16* __restrict__ QS,
                                              const u16* __restrict__ QN,
                                              const u16* __restrict__ KB,
                                              const float* __restrict__ VF,
                                              const u16* __restrict__ KP,
                                              const u16* __restrict__ VPT,
                                              const float* __restrict__ TAB,
                                              const int* __restrict__ RPI,
                                              const float* __restrict__ RPB,
                                              const float* __restrict__ LT,
                                              const float* __restrict__ LB,
                                              u16* __restrict__ Y) {
  __shared__ alignas(16) u16 sQ[64 * 56];
  __shared__ alignas(16) u16 sKp[64 * 56];
  __shared__ alignas(16) u16 sV[32 * 72];
  __shared__ alignas(16) float sAttn[64 * 68];
  __shared__ alignas(16) u16 sP[64 * 72];
  __shared__ alignas(16) float sXp[64 * 36];

  const int t = threadIdx.x;
  const int lane = t & 31;
  const int wave = t >> 5;               // 0..3, 16 output rows each
  const int mt = blockIdx.x;             // 0..48 (64-row tiles)
  const int h = blockIdx.y, b = blockIdx.z;
  const int bh = b * NHEAD + h;
  const int n0 = mt * 64;

  { // load q_s tile 64x32 bf16
    int r = t >> 1, part = (t & 1) * 16;
    const int4* src = (const int4*)(QS + ((size_t)bh * NTOK + n0 + r) * 32 + part);
    int4* dst = (int4*)(sQ + r * 56 + part);
    dst[0] = src[0]; dst[1] = src[1];
  }
  { // load k_pool tile 64x32 bf16 (already [p][d] == B-fragment layout)
    int r = t >> 1, part = (t & 1) * 16;
    const int4* src = (const int4*)(KP + ((size_t)bh * 64 + r) * 32 + part);
    int4* dst = (int4*)(sKp + r * 56 + part);
    dst[0] = src[0]; dst[1] = src[1];
  }
  { // load v_pool^T tile 32x64 bf16 ([d][p] == B-fragment layout)
    int d = t >> 2, part = (t & 3) * 16;
    const int4* src = (const int4*)(VPT + ((size_t)bh * 32 + d) * 64 + part);
    int4* dst = (int4*)(sV + d * 72 + part);
    dst[0] = src[0];
  }
  __syncthreads();

  // ---- WMMA 1: attn_pool[64 x 64] = Qs(64x32) @ Kp^T(32x64), + pool bias ----
  {
    v16bf afr = load_frag_A(sQ, 56, wave * 16, 0, lane);
    const int hl = lane >> 4, ln = lane & 15;
    #pragma unroll
    for (int ni = 0; ni < 4; ni++) {
      v8f acc = (v8f){0.f,0.f,0.f,0.f,0.f,0.f,0.f,0.f};
      v16bf bfr = load_frag_B(sKp, 56, ni * 16, 0, lane);
      acc = __builtin_amdgcn_wmma_f32_16x16x32_bf16(false, afr, false, bfr,
                                                    (short)0, acc, false, false);
      #pragma unroll
      for (int g = 0; g < 8; g++) {
        int row = wave * 16 + g + 8 * hl;
        int m = ni * 16 + ln;
        float v;
        if (m < PLEN) v = acc[g] + TAB[RPI[(size_t)(n0 + row) * PLEN + m] * 8 + h];
        else          v = NEGV;
        sAttn[row * 68 + m] = v;
      }
    }
  }
  __syncthreads();

  // ---- local window logits + softmax(9 + 49) + learnable tokens + x_local ----
  float xl[32];
  const int r = t;                       // row id for t < 64
  const int n = n0 + r;
  if (t < 64) {
    int iy = n / HW, ix = n % HW;
    float qf[32];
    const u32* qr = (const u32*)(sQ + r * 56);
    #pragma unroll
    for (int i = 0; i < 16; i++) {
      u32 u = qr[i];
      qf[2 * i] = bf2f((u16)(u & 0xffff));
      qf[2 * i + 1] = bf2f((u16)(u >> 16));
    }
    float aloc[9];
    #pragma unroll
    for (int wj = 0; wj < 9; wj++) {
      int dy = wj / 3 - 1, dx = wj % 3 - 1;
      int ny = iy + dy, nx = ix + dx;
      float s = NEGV;
      if (ny >= 0 && ny < HW && nx >= 0 && nx < HW) {
        const u32* kr = (const u32*)(KB + ((size_t)bh * NTOK + ny * HW + nx) * 32);
        float d0 = 0.f;
        #pragma unroll
        for (int i = 0; i < 16; i++) {
          u32 u = kr[i];
          d0 += qf[2 * i] * bf2f((u16)(u & 0xffff));
          d0 += qf[2 * i + 1] * bf2f((u16)(u >> 16));
        }
        s = d0 + RPB[h * 9 + wj];
      }
      aloc[wj] = s;
    }
    // softmax over [local 9 | pool 49]
    float mx = -1e30f;
    #pragma unroll
    for (int wj = 0; wj < 9; wj++) mx = fmaxf(mx, aloc[wj]);
    float* arow = sAttn + r * 68;
    for (int m = 0; m < PLEN; m++) mx = fmaxf(mx, arow[m]);
    float sum = 0.f;
    #pragma unroll
    for (int wj = 0; wj < 9; wj++) { aloc[wj] = expf(aloc[wj] - mx); sum += aloc[wj]; }
    for (int m = 0; m < PLEN; m++) { float e = expf(arow[m] - mx); arow[m] = e; sum += e; }
    float inv = 1.0f / sum;
    for (int m = 0; m < 64; m++)
      sP[r * 72 + m] = (m < PLEN) ? f2bf(arow[m] * inv) : (u16)0;
    #pragma unroll
    for (int wj = 0; wj < 9; wj++) aloc[wj] *= inv;
    // add learnable-token term (uses q_norm) to local probs
    float qnf[32];
    const u32* qnr = (const u32*)(QN + ((size_t)bh * NTOK + n) * 32);
    #pragma unroll
    for (int i = 0; i < 16; i++) {
      u32 u = qnr[i];
      qnf[2 * i] = bf2f((u16)(u & 0xffff));
      qnf[2 * i + 1] = bf2f((u16)(u >> 16));
    }
    #pragma unroll
    for (int wj = 0; wj < 9; wj++) {
      float a2 = LB[h * 9 + wj];
      #pragma unroll
      for (int d = 0; d < 32; d++) a2 += qnf[d] * LT[(h * 32 + d) * 9 + wj];
      aloc[wj] += a2;
    }
    // x_local (skip invalid taps: reference zero-masks v there)
    #pragma unroll
    for (int d = 0; d < 32; d++) xl[d] = 0.f;
    #pragma unroll
    for (int wj = 0; wj < 9; wj++) {
      int dy = wj / 3 - 1, dx = wj % 3 - 1;
      int ny = iy + dy, nx = ix + dx;
      if (ny >= 0 && ny < HW && nx >= 0 && nx < HW) {
        const float* vr = VF + ((size_t)bh * NTOK + ny * HW + nx) * 32;
        float a2 = aloc[wj];
        #pragma unroll
        for (int d = 0; d < 32; d++) xl[d] += a2 * vr[d];
      }
    }
  }
  __syncthreads();

  // ---- WMMA 2: x_pool[64 x 32] = a_pool(64x64) @ v_pool(64x32) ----
  {
    const int hl = lane >> 4, ln = lane & 15;
    v8f acc2[2];
    #pragma unroll
    for (int ni = 0; ni < 2; ni++) acc2[ni] = (v8f){0.f,0.f,0.f,0.f,0.f,0.f,0.f,0.f};
    #pragma unroll
    for (int kk = 0; kk < 2; kk++) {
      v16bf afr = load_frag_A(sP, 72, wave * 16, kk * 32, lane);
      #pragma unroll
      for (int ni = 0; ni < 2; ni++) {
        v16bf bfr = load_frag_B(sV, 72, ni * 16, kk * 32, lane);
        acc2[ni] = __builtin_amdgcn_wmma_f32_16x16x32_bf16(false, afr, false, bfr,
                                                           (short)0, acc2[ni], false, false);
      }
    }
    #pragma unroll
    for (int ni = 0; ni < 2; ni++)
      #pragma unroll
      for (int g = 0; g < 8; g++)
        sXp[(wave * 16 + g + 8 * hl) * 36 + ni * 16 + ln] = acc2[ni][g];
  }
  __syncthreads();

  // ---- combine + store y (bf16, [b][n][h*32+d]) ----
  if (t < 64) {
    u32* yb = (u32*)(Y + ((size_t)(b * NTOK + n)) * DIMC + h * HD);
    #pragma unroll
    for (int d = 0; d < 32; d += 2) {
      float a = xl[d] + sXp[r * 36 + d];
      float c = xl[d + 1] + sXp[r * 36 + d + 1];
      yb[d >> 1] = (u32)f2bf(a) | ((u32)f2bf(c) << 16);
    }
  }
}

// ---------------- launcher ----------------
extern "C" void kernel_launch(void* const* d_in, const int* in_sizes, int n_in,
                              void* d_out, int out_size, void* d_ws, size_t ws_size,
                              hipStream_t stream) {
  const float* x     = (const float*)d_in[0];
  const float* sls   = (const float*)d_in[1];
  const float* rct   = (const float*)d_in[2];
  const float* qw    = (const float*)d_in[3];
  const float* qb    = (const float*)d_in[4];
  const float* qe    = (const float*)d_in[5];
  const float* temp  = (const float*)d_in[6];
  const float* kvw   = (const float*)d_in[7];
  const float* kvb   = (const float*)d_in[8];
  const float* srw   = (const float*)d_in[9];
  const float* srb   = (const float*)d_in[10];
  const float* ng    = (const float*)d_in[11];
  const float* nb    = (const float*)d_in[12];
  const float* fc1w  = (const float*)d_in[13];
  const float* fc1b  = (const float*)d_in[14];
  const float* fc2w  = (const float*)d_in[15];
  const float* fc2b  = (const float*)d_in[16];
  const float* rpb   = (const float*)d_in[17];
  const float* lt    = (const float*)d_in[18];
  const float* lb    = (const float*)d_in[19];
  const float* projw = (const float*)d_in[20];
  const float* projb = (const float*)d_in[21];
  const int*   rpi   = (const int*)d_in[22];

  char* ws = (char*)d_ws;
  u16*   xbf   = (u16*)(ws + OFF_XBF);
  u16*   wcat  = (u16*)(ws + OFF_WCAT);
  float* biac  = (float*)(ws + OFF_BIA);
  u16*   prjbf = (u16*)(ws + OFF_PRJ);
  float* tab   = (float*)(ws + OFF_TAB);
  float* xpln  = (float*)(ws + OFF_XPL);
  u16*   kp    = (u16*)(ws + OFF_KP);
  u16*   vpt   = (u16*)(ws + OFF_VPT);
  u16*   qsb   = (u16*)(ws + OFF_QS);
  u16*   qnb   = (u16*)(ws + OFF_QN);
  u16*   kbf   = (u16*)(ws + OFF_KBF);
  float* vf    = (float*)(ws + OFF_VF);
  float* out1  = (float*)(ws + OFF_OUT1);
  u16*   ybf   = (u16*)(ws + OFF_YBF);
  float* out   = (float*)d_out;

  k_pack<<<1024, 256, 0, stream>>>(qw, kvw, srw, qb, kvb, srb, projw,
                                   wcat, biac, prjbf, kp, vpt);
  k_cvt_x<<<M1, 256, 0, stream>>>(x, xbf);                      // M1*256 elems
  k_gemm_bf16<<<dim3(M1 / 128, NCAT / 128), 256, 0, stream>>>(xbf, wcat, biac, out1, NCAT);
  k_epilogue1<<<M1, 256, 0, stream>>>(out1, qe, temp, sls, qsb, qnb, kbf, vf);
  k_poolln<<<BATCH * PLEN, 256, 0, stream>>>(out1, ng, nb, xpln);
  k_kvpool<<<BATCH * PLEN, 256, 0, stream>>>(xpln, kvw, kvb, kp, vpt);
  k_cpb<<<512, 256, 0, stream>>>(rct, fc1w, fc1b, fc2w, fc2b, tab);
  k_attn<<<dim3(NTOK / 64, NHEAD, BATCH), 128, 0, stream>>>(qsb, qnb, kbf, vf, kp, vpt,
                                                            tab, rpi, rpb, lt, lb, ybf);
  k_gemm_bf16<<<dim3(M1 / 128, DIMC / 128), 256, 0, stream>>>(ybf, prjbf, projb, out, DIMC);
}
